// BiDAF_75840532512849
// MI455X (gfx1250) — compile-verified, matrix operand order checked by reading, not run on hardware
//
#include <hip/hip_runtime.h>
#include <math.h>

typedef _Float16 f16;
typedef __attribute__((ext_vector_type(16))) _Float16 v16h;
typedef __attribute__((ext_vector_type(8)))  _Float16 v8h;
typedef __attribute__((ext_vector_type(8)))  float    v8f;

#define Hh   256
#define Bb   32
#define CLEN 400
#define QLEN 64
#define G4H  1024
#define D2H  512
#define D8H  2048
#define BC   (Bb*CLEN)   // 12800
#define BQ   (Bb*QLEN)   // 2048

// k_lstm LDS layout (dynamic): [h: B*H f16][z: B*4H f32][pre stage: B*4H f16]
#define LDS_H_OFF   0
#define LDS_Z_OFF   (Bb*Hh*2)                  // 16384
#define LDS_PRE_OFF (LDS_Z_OFF + Bb*G4H*4)     // 147456
#define LDS_TOTAL   (LDS_PRE_OFF + Bb*G4H*2)   // 212992

__device__ __forceinline__ float sigf(float x) { return 1.0f / (1.0f + __expf(-x)); }

// ---------------- embedding gather (fp32 table -> f16 activations) ----------------
__global__ void k_embed(const float* __restrict__ emb, const int* __restrict__ words,
                        f16* __restrict__ out, int total) {
    int i = blockIdx.x * blockDim.x + threadIdx.x;
    if (i >= total) return;
    int row = i >> 8;           // / H
    int d   = i & 255;
    out[i] = (f16)emb[(size_t)words[row] * Hh + d];
}

// ---------------- fp32 -> f16 convert ----------------
__global__ void k_f32tof16(const float* __restrict__ in, f16* __restrict__ out, int n) {
    int i = blockIdx.x * blockDim.x + threadIdx.x;
    if (i < n) out[i] = (f16)in[i];
}

// ---------------- generic batched WMMA GEMM:  Out[M,N] = A[M,K] * W[N,K]^T + bias ----------------
// one wave per 16x64 output tile (4 N-subtiles share one A fragment -> 4x less A traffic)
template<bool OUT_F16>
__global__ void k_gemm(const f16* __restrict__ A, const f16* __restrict__ W,
                       const float* __restrict__ bias, void* __restrict__ Out,
                       int M, int N, int K,
                       long long sA, long long sW, long long sO) {
    int wave = blockIdx.x * (blockDim.x >> 5) + (threadIdx.x >> 5);
    int tilesN = N >> 6;                      // 64-wide N tiles
    int totalTiles = (M >> 4) * tilesN;
    if (wave >= totalTiles) return;
    int tm = wave / tilesN, tn4 = wave % tilesN;
    int lane = threadIdx.x & 31;
    int r15 = lane & 15;
    int kh  = (lane >> 4) << 4;               // K sub-block 0 or 16 (contiguous k = 16h + s, same for A and B)
    const f16* Ap = A + (long long)blockIdx.z * sA + (long long)(tm * 16 + r15) * K + kh;
    const f16* Wp = W + (long long)blockIdx.z * sW + (long long)(tn4 * 64 + r15) * K + kh;
    v8f acc0 = {}, acc1 = {}, acc2 = {}, acc3 = {};
    for (int k = 0; k < K; k += 32) {
        v16h a  = *(const v16h*)(Ap + k);
        v16h b0 = *(const v16h*)(Wp + k);
        v16h b1 = *(const v16h*)(Wp + (long long)16 * K + k);
        v16h b2 = *(const v16h*)(Wp + (long long)32 * K + k);
        v16h b3 = *(const v16h*)(Wp + (long long)48 * K + k);
        acc0 = __builtin_amdgcn_wmma_f32_16x16x32_f16(false, a, false, b0, (short)0, acc0, false, false);
        acc1 = __builtin_amdgcn_wmma_f32_16x16x32_f16(false, a, false, b1, (short)0, acc1, false, false);
        acc2 = __builtin_amdgcn_wmma_f32_16x16x32_f16(false, a, false, b2, (short)0, acc2, false, false);
        acc3 = __builtin_amdgcn_wmma_f32_16x16x32_f16(false, a, false, b3, (short)0, acc3, false, false);
    }
    long long ob = (long long)blockIdx.z * sO;
    v8f accs[4] = { acc0, acc1, acc2, acc3 };
#pragma unroll
    for (int nn = 0; nn < 4; ++nn) {
        int col = tn4 * 64 + nn * 16 + r15;
        float bv = bias ? bias[col] : 0.0f;
#pragma unroll
        for (int v = 0; v < 8; ++v) {
            int row = tm * 16 + ((lane >> 4) << 3) + v;   // C/D frag: VGPR v + lane-half*8
            float val = accs[nn][v] + bv;
            if (OUT_F16) ((f16*)Out)[ob + (long long)row * N + col] = (f16)val;
            else         ((float*)Out)[ob + (long long)row * N + col] = val;
        }
    }
}

// ---------------- persistent BiLSTM recurrence ----------------
// grid.x = 2 (fwd/bwd), 1024 threads = 32 waves; each wave owns exactly one 16x64 tile of
// z = h@Whh^T. h/z/pre-stage in LDS, cell state in VGPRs across all T steps.
// pre[t] gate rows are DMA'd to LDS via GLOBAL_LOAD_ASYNC_TO_LDS_B128 (ASYNCcnt) issued
// before the WMMA phase, hiding the 64KB/step read behind the GEMM; Whh streams from L2.
__global__ __launch_bounds__(1024) void k_lstm(const f16* __restrict__ pre,   // [2][B,T,4H]
                                               const f16* __restrict__ whh,   // [2][4H,H]
                                               f16* __restrict__ out,         // [B,T,2H]
                                               int T) {
    extern __shared__ char smem[];
    f16*   h_s   = (f16*)(smem + LDS_H_OFF);    // B*H halves  (16 KB)
    float* z_s   = (float*)(smem + LDS_Z_OFF);  // B*4H floats (128 KB)
    f16*   pre_s = (f16*)(smem + LDS_PRE_OFF);  // B*4H halves (64 KB)
    const int dir = blockIdx.x;
    const f16* preD = pre + (long long)dir * Bb * T * G4H;
    const f16* whhD = whh + (long long)dir * G4H * Hh;
    int tid = threadIdx.x;
    for (int i = tid; i < Bb * Hh; i += 1024) h_s[i] = (f16)0.0f;
    float cst[8];
#pragma unroll
    for (int r = 0; r < 8; ++r) cst[r] = 0.0f;
    __syncthreads();

    const int wave = tid >> 5;       // 0..31  == tile id
    const int lane = tid & 31;
    const int r15  = lane & 15;
    const int kh   = (lane >> 4) << 4;
    const int tm   = wave >> 4;      // 0..1   (M tile)
    const int tn4  = wave & 15;      // 0..15  (64-wide N tile)
    const f16* hp = h_s + (tm * 16 + r15) * Hh + kh;
    const f16* wp = whhD + (long long)(tn4 * 64 + r15) * Hh + kh;
    // phase-2 cell mapping: thread -> (batch b, 8 contiguous hidden units j0..j0+7)
    const int b2  = tid >> 5;
    const int j0  = (tid & 31) << 3;
    // this thread's slice of the pre staging buffer (LDS byte offset, gate stride 512B)
    const unsigned pre_loff = (unsigned)LDS_PRE_OFF + (unsigned)((b2 * G4H + j0) * 2);

    for (int t = 0; t < T; ++t) {
        int tt = dir ? (T - 1 - t) : t;
        long long rowb = (long long)(b2 * T + tt);
        asm volatile("" ::: "memory");   // keep Whh loads inside the time loop (stream from L2)
        // ---- async DMA: stage pre[tt] gate rows into LDS while the GEMM runs ----
        {
            unsigned long long ga = (unsigned long long)(preD + rowb * G4H + j0);
            asm volatile(
                "global_load_async_to_lds_b128 %0, %1, off\n\t"
                "global_load_async_to_lds_b128 %0, %1, off offset:512\n\t"
                "global_load_async_to_lds_b128 %0, %1, off offset:1024\n\t"
                "global_load_async_to_lds_b128 %0, %1, off offset:1536"
                :: "v"(pre_loff), "v"(ga) : "memory");
        }
        // ---- phase 1: one 16x64 WMMA tile per wave ----
        v8f acc0 = {}, acc1 = {}, acc2 = {}, acc3 = {};
#pragma unroll 2
        for (int k = 0; k < Hh; k += 32) {
            v16h a  = *(const v16h*)(hp + k);
            v16h b0 = *(const v16h*)(wp + k);
            v16h b1 = *(const v16h*)(wp + 16 * Hh + k);
            v16h b2 = *(const v16h*)(wp + 32 * Hh + k);
            v16h b3 = *(const v16h*)(wp + 48 * Hh + k);
            acc0 = __builtin_amdgcn_wmma_f32_16x16x32_f16(false, a, false, b0, (short)0, acc0, false, false);
            acc1 = __builtin_amdgcn_wmma_f32_16x16x32_f16(false, a, false, b1, (short)0, acc1, false, false);
            acc2 = __builtin_amdgcn_wmma_f32_16x16x32_f16(false, a, false, b2, (short)0, acc2, false, false);
            acc3 = __builtin_amdgcn_wmma_f32_16x16x32_f16(false, a, false, b3, (short)0, acc3, false, false);
        }
        {
            v8f accs[4] = { acc0, acc1, acc2, acc3 };
#pragma unroll
            for (int nn = 0; nn < 4; ++nn) {
                int col = tn4 * 64 + nn * 16 + r15;
#pragma unroll
                for (int v = 0; v < 8; ++v) {
                    int row = tm * 16 + ((lane >> 4) << 3) + v;
                    z_s[row * G4H + col] = accs[nn][v];
                }
            }
        }
        __syncthreads();
        // ---- phase 2: gates from LDS, vectorized 8 units/thread (i,f,g,o order) ----
        {
            asm volatile("s_wait_asynccnt 0x0" ::: "memory");   // pre[tt] stage complete
            const v8h* pp = (const v8h*)(pre_s + b2 * G4H + j0);
            const v8f* zp = (const v8f*)(z_s + b2 * G4H + j0);
            v8h pI = pp[0], pF = pp[32], pG = pp[64], pO = pp[96];   // gate stride 256 halves
            v8f zI = zp[0], zF = zp[32], zG = zp[64], zO = zp[96];   // gate stride 256 floats
            v8h hv;
#pragma unroll
            for (int i = 0; i < 8; ++i) {
                float zi = zI[i] + (float)pI[i];
                float zf = zF[i] + (float)pF[i];
                float zg = zG[i] + (float)pG[i];
                float zo = zO[i] + (float)pO[i];
                float c2 = sigf(zf) * cst[i] + sigf(zi) * tanhf(zg);
                float h2 = sigf(zo) * tanhf(c2);
                cst[i] = c2;
                hv[i] = (f16)h2;
            }
            *(v8h*)(h_s + b2 * Hh + j0) = hv;
            *(v8h*)(out + rowb * D2H + dir * Hh + j0) = hv;
        }
        __syncthreads();
    }
}

// ---------------- small attention helpers ----------------
__global__ void k_rowdot(const f16* __restrict__ X, const float* __restrict__ w,
                         const float* __restrict__ b, float* __restrict__ out,
                         int rows, int D) {
    int r = blockIdx.x * blockDim.x + threadIdx.x;
    if (r >= rows) return;
    const f16* x = X + (long long)r * D;
    float acc = b[0];
    for (int d = 0; d < D; ++d) acc += (float)x[d] * w[d];
    out[r] = acc;
}

__global__ void k_scale_cq(const f16* __restrict__ c, const float* __restrict__ wcq,
                           f16* __restrict__ ct, long long total) {
    long long i = blockIdx.x * (long long)blockDim.x + threadIdx.x;
    if (i >= total) return;
    int d = (int)(i & 511);
    ct[i] = (f16)((float)c[i] * wcq[d]);
}

__global__ void k_qT(const f16* __restrict__ q, f16* __restrict__ qT) {
    int i = blockIdx.x * blockDim.x + threadIdx.x;     // B*Q*2H
    if (i >= BQ * D2H) return;
    int b  = i / (QLEN * D2H);
    int rem = i % (QLEN * D2H);
    int qq = rem / D2H;
    int d  = rem % D2H;
    qT[(long long)b * D2H * QLEN + d * QLEN + qq] = q[i];
}

// one wave per (b,c) row: add rank-1 terms, softmax over Q=64, emit a (f16) and row max
__global__ void k_softmax_s(const float* __restrict__ s, const float* __restrict__ cw,
                            const float* __restrict__ qw, const float* __restrict__ bcq,
                            f16* __restrict__ a, float* __restrict__ smax) {
    int row = blockIdx.x * (blockDim.x >> 5) + (threadIdx.x >> 5);
    if (row >= BC) return;
    int lane = threadIdx.x & 31;
    int b = row / CLEN;
    float base = cw[row] + bcq[0];
    float v0 = s[(long long)row * QLEN + lane]      + base + qw[b * QLEN + lane];
    float v1 = s[(long long)row * QLEN + lane + 32] + base + qw[b * QLEN + lane + 32];
    float m = fmaxf(v0, v1);
    for (int off = 16; off; off >>= 1) m = fmaxf(m, __shfl_xor(m, off, 32));
    float e0 = __expf(v0 - m), e1 = __expf(v1 - m);
    float sum = e0 + e1;
    for (int off = 16; off; off >>= 1) sum += __shfl_xor(sum, off, 32);
    float inv = 1.0f / sum;
    a[(long long)row * QLEN + lane]      = (f16)(e0 * inv);
    a[(long long)row * QLEN + lane + 32] = (f16)(e1 * inv);
    if (lane == 0) smax[row] = m;
}

// softmax over C of row-maxes, then q2c[b,d] = sum_c bwt[c] * c[b,c,d]
__global__ __launch_bounds__(512) void k_q2c(const float* __restrict__ smax,
                                             const f16* __restrict__ c,
                                             float* __restrict__ q2c) {
    __shared__ float bw[CLEN];
    __shared__ float norm;
    int b = blockIdx.x;
    int tid = threadIdx.x;
    for (int i = tid; i < CLEN; i += 512) bw[i] = smax[b * CLEN + i];
    __syncthreads();
    if (tid == 0) {
        float m = -1e30f;
        for (int i = 0; i < CLEN; ++i) m = fmaxf(m, bw[i]);
        float s = 0.0f;
        for (int i = 0; i < CLEN; ++i) { bw[i] = __expf(bw[i] - m); s += bw[i]; }
        norm = 1.0f / s;
    }
    __syncthreads();
    float inv = norm;
    float acc = 0.0f;
    for (int cp = 0; cp < CLEN; ++cp)
        acc += bw[cp] * (float)c[((long long)(b * CLEN + cp)) * D2H + tid];
    q2c[b * D2H + tid] = acc * inv;
}

__global__ void k_buildg(const f16* __restrict__ c, const f16* __restrict__ c2q,
                         const float* __restrict__ q2c, f16* __restrict__ g) {
    long long i = blockIdx.x * (long long)blockDim.x + threadIdx.x;
    if (i >= (long long)BC * D2H) return;
    long long rowd = i >> 9;
    int d = (int)(i & 511);
    int b = (int)(rowd / CLEN);
    float cv   = (float)c[i];
    float c2qv = (float)c2q[i];
    float q2cv = q2c[b * D2H + d];
    f16* gr = g + rowd * D8H;
    gr[d]        = (f16)cv;
    gr[512 + d]  = (f16)c2qv;
    gr[1024 + d] = (f16)(cv * c2qv);
    gr[1536 + d] = (f16)(cv * q2cv);
}

__global__ void k_final(const f16* __restrict__ g, const f16* __restrict__ m,
                        const f16* __restrict__ m2,
                        const float* p1wg, const float* p1wm,
                        const float* p2wg, const float* p2wm,
                        const float* p1bg, const float* p1bm,
                        const float* p2bg, const float* p2bm,
                        float* __restrict__ out) {
    int rowd = blockIdx.x * blockDim.x + threadIdx.x;
    if (rowd >= BC) return;
    const f16* gr  = g  + (long long)rowd * D8H;
    const f16* mr  = m  + (long long)rowd * D2H;
    const f16* m2r = m2 + (long long)rowd * D2H;
    float a1 = p1bg[0] + p1bm[0];
    float a2 = p2bg[0] + p2bm[0];
    for (int d = 0; d < D8H; ++d) { float gv = (float)gr[d]; a1 += gv * p1wg[d]; a2 += gv * p2wg[d]; }
    for (int d = 0; d < D2H; ++d) { a1 += (float)mr[d] * p1wm[d]; a2 += (float)m2r[d] * p2wm[d]; }
    out[rowd]      = a1;
    out[BC + rowd] = a2;
}

// =========================================================================
extern "C" void kernel_launch(void* const* d_in, const int* in_sizes, int n_in,
                              void* d_out, int out_size, void* d_ws, size_t ws_size,
                              hipStream_t stream) {
    (void)in_sizes; (void)n_in; (void)out_size; (void)ws_size;
    // input indices per setup_inputs() dict order
    const float* emb = (const float*)d_in[0];
    const float* W_wih[3][2] = { {(const float*)d_in[1],  (const float*)d_in[4]},
                                 {(const float*)d_in[7],  (const float*)d_in[10]},
                                 {(const float*)d_in[13], (const float*)d_in[16]} };
    const float* W_whh[3][2] = { {(const float*)d_in[2],  (const float*)d_in[5]},
                                 {(const float*)d_in[8],  (const float*)d_in[11]},
                                 {(const float*)d_in[14], (const float*)d_in[17]} };
    const float* Bias[3][2]  = { {(const float*)d_in[3],  (const float*)d_in[6]},
                                 {(const float*)d_in[9],  (const float*)d_in[12]},
                                 {(const float*)d_in[15], (const float*)d_in[18]} };
    const float* att_w_c  = (const float*)d_in[19];
    const float* att_w_q  = (const float*)d_in[20];
    const float* att_w_cq = (const float*)d_in[21];
    const float* att_b_c  = (const float*)d_in[22];
    const float* att_b_q  = (const float*)d_in[23];
    const float* att_b_cq = (const float*)d_in[24];
    const float* p1_wg = (const float*)d_in[25];
    const float* p1_wm = (const float*)d_in[26];
    const float* p2_wg = (const float*)d_in[27];
    const float* p2_wm = (const float*)d_in[28];
    const float* p1_bg = (const float*)d_in[29];
    const float* p1_bm = (const float*)d_in[30];
    const float* p2_bg = (const float*)d_in[31];
    const float* p2_bm = (const float*)d_in[32];
    const int* c_word = (const int*)d_in[33];
    const int* q_word = (const int*)d_in[34];

    size_t cur = 0;
    auto alloc = [&](size_t bytes) -> void* {
        cur = (cur + 255) & ~(size_t)255;
        void* p = (char*)d_ws + cur;
        cur += bytes;
        return p;
    };
    f16* cemb16 = (f16*)alloc((size_t)BC * Hh * 2);
    f16* qemb16 = (f16*)alloc((size_t)BQ * Hh * 2);
    int wih_k[3] = { Hh, D8H, D2H };
    f16* wih16[3]; f16* whh16[3];
    for (int l = 0; l < 3; ++l) {
        wih16[l] = (f16*)alloc((size_t)2 * G4H * wih_k[l] * 2);
        whh16[l] = (f16*)alloc((size_t)2 * G4H * Hh * 2);
    }
    f16* pre    = (f16*)alloc((size_t)2 * BC * G4H * 2);   // reused: ctx-c, mod, out
    f16* pre_q  = (f16*)alloc((size_t)2 * BQ * G4H * 2);
    f16* c16    = (f16*)alloc((size_t)BC * D2H * 2);
    f16* q16    = (f16*)alloc((size_t)BQ * D2H * 2);
    f16* qT16   = (f16*)alloc((size_t)BQ * D2H * 2);
    float* cw   = (float*)alloc((size_t)BC * 4);
    float* qw   = (float*)alloc((size_t)BQ * 4);
    float* smax = (float*)alloc((size_t)BC * 4);
    f16* ct16   = (f16*)alloc((size_t)BC * D2H * 2);
    float* s32  = (float*)alloc((size_t)BC * QLEN * 4);
    f16* a16    = (f16*)alloc((size_t)BC * QLEN * 2);
    f16* c2q16  = (f16*)alloc((size_t)BC * D2H * 2);
    float* q2c  = (float*)alloc((size_t)Bb * D2H * 4);
    f16* g16    = (f16*)alloc((size_t)BC * D8H * 2);
    f16* m16    = (f16*)alloc((size_t)BC * D2H * 2);
    f16* m2_16  = (f16*)alloc((size_t)BC * D2H * 2);

    // embeddings
    k_embed<<<dim3((BC * Hh + 255) / 256), dim3(256), 0, stream>>>(emb, c_word, cemb16, BC * Hh);
    k_embed<<<dim3((BQ * Hh + 255) / 256), dim3(256), 0, stream>>>(emb, q_word, qemb16, BQ * Hh);

    // weight conversion fp32 -> f16 (pack fwd then bwd contiguously)
    for (int l = 0; l < 3; ++l) {
        int nih = G4H * wih_k[l];
        int nhh = G4H * Hh;
        for (int d = 0; d < 2; ++d) {
            k_f32tof16<<<dim3((nih + 255) / 256), dim3(256), 0, stream>>>(W_wih[l][d], wih16[l] + (size_t)d * nih, nih);
            k_f32tof16<<<dim3((nhh + 255) / 256), dim3(256), 0, stream>>>(W_whh[l][d], whh16[l] + (size_t)d * nhh, nhh);
        }
    }

    auto gemm_f16 = [&](const f16* A, const f16* W, const float* bias, f16* O,
                        int M, int N, int K, int batch, long long sA, long long sW, long long sO) {
        int tiles = (M >> 4) * (N >> 6);
        k_gemm<true><<<dim3((tiles + 3) / 4, 1, batch), dim3(128), 0, stream>>>(
            A, W, bias, (void*)O, M, N, K, sA, sW, sO);
    };
    auto gemm_f32 = [&](const f16* A, const f16* W, const float* bias, float* O,
                        int M, int N, int K, int batch, long long sA, long long sW, long long sO) {
        int tiles = (M >> 4) * (N >> 6);
        k_gemm<false><<<dim3((tiles + 3) / 4, 1, batch), dim3(128), 0, stream>>>(
            A, W, bias, (void*)O, M, N, K, sA, sW, sO);
    };

    // ---- ctx BiLSTM over context and question ----
    for (int d = 0; d < 2; ++d)
        gemm_f16(cemb16, wih16[0] + (size_t)d * G4H * Hh, Bias[0][d],
                 pre + (size_t)d * BC * G4H, BC, G4H, Hh, 1, 0, 0, 0);
    k_lstm<<<dim3(2), dim3(1024), LDS_TOTAL, stream>>>(pre, whh16[0], c16, CLEN);
    for (int d = 0; d < 2; ++d)
        gemm_f16(qemb16, wih16[0] + (size_t)d * G4H * Hh, Bias[0][d],
                 pre_q + (size_t)d * BQ * G4H, BQ, G4H, Hh, 1, 0, 0, 0);
    k_lstm<<<dim3(2), dim3(1024), LDS_TOTAL, stream>>>(pre_q, whh16[0], q16, QLEN);

    // ---- BiDAF attention ----
    k_rowdot<<<dim3((BC + 255) / 256), dim3(256), 0, stream>>>(c16, att_w_c, att_b_c, cw, BC, D2H);
    k_rowdot<<<dim3((BQ + 255) / 256), dim3(256), 0, stream>>>(q16, att_w_q, att_b_q, qw, BQ, D2H);
    k_scale_cq<<<dim3((BC * D2H + 255) / 256), dim3(256), 0, stream>>>(c16, att_w_cq, ct16, (long long)BC * D2H);
    k_qT<<<dim3((BQ * D2H + 255) / 256), dim3(256), 0, stream>>>(q16, qT16);
    // trilinear: per batch [400,512] x [64,512]^T -> s[400,64]
    gemm_f32(ct16, q16, nullptr, s32, CLEN, QLEN, D2H, Bb,
             (long long)CLEN * D2H, (long long)QLEN * D2H, (long long)CLEN * QLEN);
    k_softmax_s<<<dim3((BC + 3) / 4), dim3(128), 0, stream>>>(s32, cw, qw, att_b_cq, a16, smax);
    // c2q: per batch a[400,64] x qT[512,64]^T -> [400,512]
    gemm_f16(a16, qT16, nullptr, c2q16, CLEN, D2H, QLEN, Bb,
             (long long)CLEN * QLEN, (long long)D2H * QLEN, (long long)CLEN * D2H);
    k_q2c<<<dim3(Bb), dim3(512), 0, stream>>>(smax, c16, q2c);
    k_buildg<<<dim3((BC * D2H + 255) / 256), dim3(256), 0, stream>>>(c16, c2q16, q2c, g16);

    // ---- modeling BiLSTM (K = 2048) ----
    for (int d = 0; d < 2; ++d)
        gemm_f16(g16, wih16[1] + (size_t)d * G4H * D8H, Bias[1][d],
                 pre + (size_t)d * BC * G4H, BC, G4H, D8H, 1, 0, 0, 0);
    k_lstm<<<dim3(2), dim3(1024), LDS_TOTAL, stream>>>(pre, whh16[1], m16, CLEN);

    // ---- output BiLSTM (K = 512) ----
    for (int d = 0; d < 2; ++d)
        gemm_f16(m16, wih16[2] + (size_t)d * G4H * D2H, Bias[2][d],
                 pre + (size_t)d * BC * G4H, BC, G4H, D2H, 1, 0, 0, 0);
    k_lstm<<<dim3(2), dim3(1024), LDS_TOTAL, stream>>>(pre, whh16[2], m2_16, CLEN);

    // ---- p1 / p2 projections ----
    k_final<<<dim3((BC + 255) / 256), dim3(256), 0, stream>>>(
        g16, m16, m2_16, p1_wg, p1_wm, p2_wg, p2_wm, p1_bg, p1_bm, p2_bg, p2_bm, (float*)d_out);
}
